// MambaLayer_24017457119248
// MI455X (gfx1250) — compile-verified, matrix operand order checked by reading, not run on hardware
//
#include <hip/hip_runtime.h>
#include <math.h>

typedef _Float16 h16;
typedef __attribute__((ext_vector_type(16))) _Float16 v16h;
typedef __attribute__((ext_vector_type(8)))  float    v8f;

#define NTOK   32768
#define DIMC   128
#define DINNER 256
#define NCHUNK 256
#define CLEN   128

__device__ __forceinline__ float fsilu(float x)     { return x / (1.f + __expf(-x)); }
__device__ __forceinline__ float fsoftplus(float x) { return x > 20.f ? x : log1pf(__expf(x)); }

// ---------------------------------------------------------------------------
// Weight prep: f32 -> f16 copies, padding, A = -exp(A_log)
// ---------------------------------------------------------------------------
__global__ void prep_small_kernel(const float* __restrict__ W_in, const float* __restrict__ W_x,
                                  const float* __restrict__ A_log,
                                  h16* __restrict__ Winh, h16* __restrict__ Wxh,
                                  float* __restrict__ Aneg)
{
    int i = blockIdx.x * blockDim.x + threadIdx.x;
    if (i < 512 * 128) Winh[i] = (h16)W_in[i];                 // (512 x 128) row-major (N,K)
    if (i < 48 * 256) {
        int row = i >> 8;                                      // padded (48 x 256)
        Wxh[i] = (row < 40) ? (h16)W_x[i] : (h16)0.f;
    }
    if (i < 4096) Aneg[i] = -__expf(A_log[i]);                 // (256 x 16)
}

// M_r = proj_w[:, r*128:(r+1)*128] @ W_out  -> (3, 128, 256) f16, row-major (N,K) per r
__global__ void fuse_proj_kernel(const float* __restrict__ projw, const float* __restrict__ Wout,
                                 h16* __restrict__ Mh)
{
    int idx = blockIdx.x * blockDim.x + threadIdx.x;           // 3*128*256
    int r = idx >> 15;
    int o = (idx >> 8) & 127;
    int dcol = idx & 255;
    float acc = 0.f;
#pragma unroll 8
    for (int c = 0; c < 128; ++c)
        acc += projw[o * 384 + r * 128 + c] * Wout[c * 256 + dcol];
    Mh[idx] = (h16)acc;
}

// ---------------------------------------------------------------------------
// Gather (direction permutation) + LayerNorm over 128 channels
// ---------------------------------------------------------------------------
__global__ void gather_ln_kernel(const float* __restrict__ x, const float* __restrict__ w,
                                 const float* __restrict__ b, float* __restrict__ xf, int dir)
{
    __shared__ float s1[DIMC], s2[DIMC];
    int t = blockIdx.x, c = threadIdx.x;
    int p;
    if (dir == 0)      p = t;
    else if (dir == 1) p = (t & 31) * 1024 + (t >> 10) * 32 + ((t >> 5) & 31);
    else               p = ((t >> 5) & 31) * 1024 + (t & 31) * 32 + (t >> 10);
    float v = x[(size_t)c * NTOK + p];
    s1[c] = v; s2[c] = v * v;
    __syncthreads();
    for (int st = 64; st > 0; st >>= 1) {
        if (c < st) { s1[c] += s1[c + st]; s2[c] += s2[c + st]; }
        __syncthreads();
    }
    float mu  = s1[0] * (1.f / 128.f);
    float var = s2[0] * (1.f / 128.f) - mu * mu;
    xf[(size_t)t * DIMC + c] = (v - mu) * rsqrtf(var + 1e-5f) * w[c] + b[c];
}

// ---------------------------------------------------------------------------
// WMMA GEMM: C(M,N) = A(M,K) @ W(N,K)^T ; A row-major (f32 or f16), W row-major f16.
// Block = 256 thr = 8 waves; each wave owns a 16(M) x 16*NT(N) strip: one A fragment
// per k-step feeds NT independent WMMA accumulator chains (A reuse + hazard hiding).
// 16-bit fragment layout (ISA 7.12.2): lanes 0-15 hold K 0-7 & 16-23, lanes 16-31
// hold K 8-15 & 24-31; A row / B col = lane%16.
// ---------------------------------------------------------------------------
template<typename AT>
__device__ __forceinline__ v16h load_frag(const AT* __restrict__ base, int ld, int row, int kb)
{
    v16h f;
    const AT* p = base + (size_t)row * ld + kb;
#pragma unroll
    for (int i = 0; i < 8; ++i) { f[i] = (h16)p[i]; f[i + 8] = (h16)p[i + 16]; }
    return f;
}

// EPI: 0 = split into u_raw/z (f16), 1 = plain f32 store with ldc, 2 = transposed out (+bias / accumulate)
template<int EPI, int K, int NT, typename AT>
__global__ void wmma_gemm_kernel(const AT* __restrict__ A, const h16* __restrict__ W,
                                 void* __restrict__ out0, void* __restrict__ out1,
                                 const float* __restrict__ bias, int ldc, int accum)
{
    int lane  = threadIdx.x & 31;
    int wave  = threadIdx.x >> 5;
    int mbase = blockIdx.x * 128 + wave * 16;
    int nbase = blockIdx.y * (16 * NT);
    int rA = mbase + (lane & 15);
    int nB = nbase + (lane & 15);
    int kh = (lane >> 4) * 8;
    v8f c[NT];
#pragma unroll
    for (int j = 0; j < NT; ++j) c[j] = (v8f){};
#pragma unroll
    for (int k0 = 0; k0 < K; k0 += 32) {
        v16h a = load_frag<AT>(A, K, rA, k0 + kh);
#pragma unroll
        for (int j = 0; j < NT; ++j) {
            v16h bf = load_frag<h16>(W, K, nB + j * 16, k0 + kh);
            c[j] = __builtin_amdgcn_wmma_f32_16x16x32_f16(false, a, false, bf,
                                                          (short)0, c[j], false, false);
        }
    }
    int row0 = mbase + 8 * (lane >> 4);
    if constexpr (EPI == 0) {
        h16* u = (h16*)out0; h16* z = (h16*)out1;
#pragma unroll
        for (int j = 0; j < NT; ++j) {
            int col = nbase + j * 16 + (lane & 15);
#pragma unroll
            for (int r = 0; r < 8; ++r) {
                int row = row0 + r;
                if (col < DINNER) u[(size_t)row * DINNER + col] = (h16)c[j][r];
                else              z[(size_t)row * DINNER + (col - DINNER)] = (h16)c[j][r];
            }
        }
    } else if constexpr (EPI == 1) {
        float* o = (float*)out0;
#pragma unroll
        for (int j = 0; j < NT; ++j) {
            int col = nbase + j * 16 + (lane & 15);
#pragma unroll
            for (int r = 0; r < 8; ++r) o[(size_t)(row0 + r) * ldc + col] = c[j][r];
        }
    } else {
        float* o = (float*)out0;                                // out[o_chan * NTOK + token]
#pragma unroll
        for (int j = 0; j < NT; ++j) {
            int col = nbase + j * 16 + (lane & 15);
#pragma unroll
            for (int r = 0; r < 8; ++r) {
                size_t idx = (size_t)col * NTOK + (row0 + r);
                float prev = accum ? o[idx] : bias[col];
                o[idx] = prev + c[j][r];
            }
        }
    }
}

// ---------------------------------------------------------------------------
// Depthwise causal conv (width 4) + SiLU : f16 in, f32 out
// ---------------------------------------------------------------------------
__global__ void conv_silu_kernel(const h16* __restrict__ u_raw, const float* __restrict__ cw,
                                 const float* __restrict__ cb, float* __restrict__ u)
{
    int idx = blockIdx.x * blockDim.x + threadIdx.x;           // NTOK*256
    int t = idx >> 8, ch = idx & 255;
    float acc = cb[ch];
#pragma unroll
    for (int k = 0; k < 4; ++k) {
        int tt = t - 3 + k;
        if (tt >= 0) acc += cw[ch * 4 + k] * (float)u_raw[(size_t)tt * DINNER + ch];
    }
    u[idx] = fsilu(acc);
}

// dt = softplus(dt_raw @ W_dt^T + b_dt), K=8 (VALU dot, not worth WMMA padding)
__global__ void dt_kernel(const float* __restrict__ xdbl, const float* __restrict__ Wdt,
                          const float* __restrict__ bdt, float* __restrict__ dt)
{
    int idx = blockIdx.x * blockDim.x + threadIdx.x;           // NTOK*256
    int t = idx >> 8, d = idx & 255;
    float acc = bdt[d];
#pragma unroll
    for (int r = 0; r < 8; ++r) acc += xdbl[(size_t)t * 48 + r] * Wdt[d * 8 + r];
    dt[idx] = fsoftplus(acc);
}

// ---------------------------------------------------------------------------
// Chunked selective scan. Lane = (channel pair within wave, state s): wave32 = 2 ch x 16 st.
// Pass 1: per-chunk summaries P = prod(a), F = local final state (h0 = 0).
// ---------------------------------------------------------------------------
__global__ void scan1_kernel(const float* __restrict__ dt, const float* __restrict__ u,
                             const float* __restrict__ xdbl, const float* __restrict__ Aneg,
                             float* __restrict__ Pb, float* __restrict__ Fb)
{
    int lane = threadIdx.x & 31, wave = threadIdx.x >> 5;
    int d = blockIdx.y * 16 + wave * 2 + (lane >> 4);
    int s = lane & 15;
    int t0 = blockIdx.x * CLEN;
    float Ads = Aneg[d * 16 + s];
    float h = 0.f, P = 1.f;
    for (int i = 0; i < CLEN; ++i) {
        int t = t0 + i;
        float dtv = dt[(size_t)t * DINNER + d];
        float uv  = u[(size_t)t * DINNER + d];
        float Bv  = xdbl[(size_t)t * 48 + 8 + s];
        float a = __expf(dtv * Ads);
        P *= a;
        h = a * h + dtv * uv * Bv;
    }
    int pi = blockIdx.x * 4096 + d * 16 + s;
    Pb[pi] = P; Fb[pi] = h;
}

// Sequential sweep over 256 chunk summaries per (d,s) pair: carry[k+1] = P_k*carry_k + F_k
__global__ void carry_kernel(const float* __restrict__ Pb, const float* __restrict__ Fb,
                             float* __restrict__ Cb)
{
    int pair = blockIdx.x * blockDim.x + threadIdx.x;          // 4096
    float c = 0.f;
    Cb[pair] = 0.f;
    for (int k = 0; k < NCHUNK - 1; ++k) {
        c = Pb[(size_t)k * 4096 + pair] * c + Fb[(size_t)k * 4096 + pair];
        Cb[(size_t)(k + 1) * 4096 + pair] = c;
    }
}

// Pass 2: replay chunk with injected carry, reduce y over states, fuse +u*Dp and *silu(z)
__global__ void scan2_kernel(const float* __restrict__ dt, const float* __restrict__ u,
                             const float* __restrict__ xdbl, const float* __restrict__ Aneg,
                             const float* __restrict__ Cb, const h16* __restrict__ zb,
                             const float* __restrict__ Dp, h16* __restrict__ yfull)
{
    int lane = threadIdx.x & 31, wave = threadIdx.x >> 5;
    int d = blockIdx.y * 16 + wave * 2 + (lane >> 4);
    int s = lane & 15;
    int t0 = blockIdx.x * CLEN;
    float Ads = Aneg[d * 16 + s];
    float h = Cb[(size_t)blockIdx.x * 4096 + d * 16 + s];
    float Dpd = Dp[d];
    for (int i = 0; i < CLEN; ++i) {
        int t = t0 + i;
        float dtv = dt[(size_t)t * DINNER + d];
        float uv  = u[(size_t)t * DINNER + d];
        float Bv  = xdbl[(size_t)t * 48 + 8 + s];
        float Cv  = xdbl[(size_t)t * 48 + 24 + s];
        float a = __expf(dtv * Ads);
        h = a * h + dtv * uv * Bv;
        float ps = h * Cv;
        ps += __shfl_xor(ps, 1, 32);                           // reduce across 16-lane state group
        ps += __shfl_xor(ps, 2, 32);
        ps += __shfl_xor(ps, 4, 32);
        ps += __shfl_xor(ps, 8, 32);
        if (s == 0) {
            float zv = (float)zb[(size_t)t * DINNER + d];
            yfull[(size_t)t * DINNER + d] = (h16)((ps + uv * Dpd) * fsilu(zv));
        }
    }
}

// ---------------------------------------------------------------------------
extern "C" void kernel_launch(void* const* d_in, const int* in_sizes, int n_in,
                              void* d_out, int out_size, void* d_ws, size_t ws_size,
                              hipStream_t stream)
{
    (void)in_sizes; (void)n_in; (void)out_size; (void)ws_size;
    const float* x      = (const float*)d_in[0];
    const float* ln_w   = (const float*)d_in[1];
    const float* ln_b   = (const float*)d_in[2];
    const float* W_in   = (const float*)d_in[3];
    const float* conv_w = (const float*)d_in[4];
    const float* conv_b = (const float*)d_in[5];
    const float* W_x    = (const float*)d_in[6];
    const float* W_dt   = (const float*)d_in[7];
    const float* b_dt   = (const float*)d_in[8];
    const float* A_log  = (const float*)d_in[9];
    const float* Dp     = (const float*)d_in[10];
    const float* W_out  = (const float*)d_in[11];
    const float* projw  = (const float*)d_in[12];
    const float* projb  = (const float*)d_in[13];
    float* out = (float*)d_out;

    char* ws = (char*)d_ws;
    size_t off = 0;
    auto alloc = [&](size_t bytes) { char* p = ws + off; off += (bytes + 255) & ~(size_t)255; return p; };

    h16*   Winh = (h16*)alloc(512 * 128 * 2);
    h16*   Wxh  = (h16*)alloc(48 * 256 * 2);
    h16*   Mh   = (h16*)alloc(3 * 128 * 256 * 2);
    float* Aneg = (float*)alloc(4096 * 4);
    float* xf   = (float*)alloc((size_t)NTOK * 128 * 4);       // aliased by yfull after GEMM1
    h16*   yfull = (h16*)xf;
    char*  scrA = alloc((size_t)NTOK * 256 * 2);               // u_raw f16, then P/F/carry after conv
    h16*   u_raw = (h16*)scrA;
    float* Pb = (float*)scrA;
    float* Fb = (float*)(scrA + (size_t)NCHUNK * 4096 * 4);
    float* Cb = (float*)(scrA + (size_t)2 * NCHUNK * 4096 * 4);
    h16*   zb   = (h16*)alloc((size_t)NTOK * 256 * 2);
    float* u    = (float*)alloc((size_t)NTOK * 256 * 4);
    float* xdbl = (float*)alloc((size_t)NTOK * 48 * 4);
    float* dtb  = (float*)alloc((size_t)NTOK * 256 * 4);

    prep_small_kernel<<<(512 * 128 + 255) / 256, 256, 0, stream>>>(W_in, W_x, A_log, Winh, Wxh, Aneg);
    fuse_proj_kernel<<<(3 * 128 * 256) / 256, 256, 0, stream>>>(projw, W_out, Mh);

    for (int dir = 0; dir < 3; ++dir) {
        gather_ln_kernel<<<NTOK, 128, 0, stream>>>(x, ln_w, ln_b, xf, dir);
        // xz = xf @ W_in^T : (32768x128)(128x512), split u_raw/z as f16; 16x64 strip per wave
        wmma_gemm_kernel<0, 128, 4, float><<<dim3(NTOK / 128, 512 / 64), 256, 0, stream>>>(
            xf, Winh, u_raw, zb, nullptr, 0, 0);
        conv_silu_kernel<<<(NTOK * 256) / 256, 256, 0, stream>>>(u_raw, conv_w, conv_b, u);
        // x_dbl = u @ W_x^T : (32768x256)(256x48 padded); 16x48 strip per wave
        wmma_gemm_kernel<1, 256, 3, float><<<dim3(NTOK / 128, 1), 256, 0, stream>>>(
            u, Wxh, xdbl, nullptr, nullptr, 48, 0);
        dt_kernel<<<(NTOK * 256) / 256, 256, 0, stream>>>(xdbl, W_dt, b_dt, dtb);
        scan1_kernel<<<dim3(NCHUNK, 16), 256, 0, stream>>>(dtb, u, xdbl, Aneg, Pb, Fb);
        carry_kernel<<<4096 / 256, 256, 0, stream>>>(Pb, Fb, Cb);
        scan2_kernel<<<dim3(NCHUNK, 16), 256, 0, stream>>>(dtb, u, xdbl, Aneg, Cb, zb, Dp, yfull);
        // out += yfull @ (proj_r @ W_out)^T  (+ proj_b on first direction)
        wmma_gemm_kernel<2, 256, 4, h16><<<dim3(NTOK / 128, 128 / 64), 256, 0, stream>>>(
            yfull, Mh + (size_t)dir * 128 * 256, out, nullptr, projb, 0, dir);
    }
}
